// DeformConv2dModule_26079041421806
// MI455X (gfx1250) — matrix operand order repaired and checked by reading, not compiled, not run on hardware
//
#include <hip/hip_runtime.h>
#include <hip/hip_bf16.h>
#include <math.h>

// ---------------- types for WMMA (gfx1250, wave32) ----------------
typedef _Float16 v16h __attribute__((ext_vector_type(16)));
typedef float    v8f  __attribute__((ext_vector_type(8)));

union FragH { uint4 u[2]; v16h v; };

// Problem constants (reference: B=4, C=256, H=W=64, Cout=256, 3x3, pad=1, stride=1)
#define BB    4
#define CC    256
#define HH    64
#define WW    64
#define COUT  256
#define KTAPS 9
#define KDIM  (CC * KTAPS)          // 2304
#define MTOT  (BB * HH * WW)        // 16384
#define GN_G  32
#define CPG   (COUT / GN_G)         // 8
#define NPG   (CPG * HH * WW)       // 32768 elements per (b,group)
#define EPSV  1e-5f

// ---------------- Kernel 1: x NCHW -> NHWC (f32) ----------------
// grid: B*H*(C/64) = 1024 blocks, 256 threads. tile = 64 channels x 64 w.
__global__ void k_transpose_x(const float* __restrict__ x, float* __restrict__ xT) {
    __shared__ float tile[64 * 65];
    const int ct = blockIdx.x & 3;
    const int h  = (blockIdx.x >> 2) & 63;
    const int b  = blockIdx.x >> 8;
    const int c0 = ct * 64;
    const int tid = threadIdx.x;
    #pragma unroll 4
    for (int it = 0; it < 16; ++it) {
        const int cc = (tid >> 6) + it * 4;
        const int w  = tid & 63;
        tile[cc * 65 + w] = x[(((size_t)(b * CC + c0 + cc) * HH + h) * WW) + w];
    }
    __syncthreads();
    #pragma unroll 4
    for (int it = 0; it < 16; ++it) {
        const int w = (tid >> 6) + it * 4;
        const int c = tid & 63;
        xT[((size_t)(b * HH * WW) + h * WW + w) * CC + c0 + c] = tile[c * 65 + w];
    }
}

// ---------------- Kernel 2: weight pack -> f16 [o][k*256+c] ----------------
__global__ void k_pack_w(const float* __restrict__ w, _Float16* __restrict__ wP) {
    const int e = blockIdx.x * 256 + threadIdx.x;
    if (e >= COUT * KDIM) return;
    const int o = e / KDIM;
    const int r = e - o * KDIM;
    const int tap = r >> 8;
    const int c   = r & 255;
    wP[(size_t)o * KDIM + r] = (_Float16)w[((size_t)(o * CC + c)) * KTAPS + tap];
}

// ---------------- Kernel 3: implicit GEMM via WMMA f16 ----------------
// 512 blocks x 256 threads (8 waves). Block: 32 pixels x all 256 outputs.
// Wave wv owns output columns [wv*32, wv*32+32).
// K loop: tap-outer (9) x 32-channel inner (8, fully unrolled, double-buffered).
__global__ void __launch_bounds__(256)
k_deform_gemm(const float* __restrict__ xT, const float* __restrict__ offset,
              const _Float16* __restrict__ wP, float* __restrict__ yws) {
    __shared__ __align__(16) float  pw[32][KTAPS][4];     // bilinear weights (validity folded)
    __shared__ __align__(16) int    pidx[32][KTAPS][4];   // clamped element offsets into xT
    __shared__ __align__(16) _Float16 ldsA[2][32 * 32];   // double-buffered A tile

    const int tid   = threadIdx.x;
    const int lane  = tid & 31;
    const int wv    = tid >> 5;          // 0..7
    const int l16   = lane & 15;
    const int khalf = lane >> 4;         // 0/1
    const int mBase = blockIdx.x * 32;
    const int b     = mBase >> 12;       // 4096 pixels per image

    // ---- precompute bilinear params: 32 pixels x 9 taps ----
    for (int i = tid; i < 32 * KTAPS; i += 256) {
        const int pix = i / KTAPS;
        const int tap = i - pix * KTAPS;
        const int m  = mBase + pix;
        const int h  = (m >> 6) & 63;
        const int w  = m & 63;
        const int ky = tap / 3, kx = tap - 3 * (tap / 3);
        const float offy = offset[((size_t)(b * 2 * KTAPS + 2 * tap)     * HH + h) * WW + w];
        const float offx = offset[((size_t)(b * 2 * KTAPS + 2 * tap + 1) * HH + h) * WW + w];
        const float py = (float)(h - 1 + ky) + offy;
        const float px = (float)(w - 1 + kx) + offx;
        const float y0f = floorf(py), x0f = floorf(px);
        const float ly = py - y0f, lx = px - x0f;
        const int y0 = (int)y0f, x0 = (int)x0f;
        const int y1 = y0 + 1,   x1 = x0 + 1;
        const float vy0 = (y0 >= 0 && y0 < HH) ? 1.f : 0.f;
        const float vy1 = (y1 >= 0 && y1 < HH) ? 1.f : 0.f;
        const float vx0 = (x0 >= 0 && x0 < WW) ? 1.f : 0.f;
        const float vx1 = (x1 >= 0 && x1 < WW) ? 1.f : 0.f;
        pw[pix][tap][0] = (1.f - ly) * (1.f - lx) * vy0 * vx0;
        pw[pix][tap][1] = (1.f - ly) * lx         * vy0 * vx1;
        pw[pix][tap][2] = ly * (1.f - lx)         * vy1 * vx0;
        pw[pix][tap][3] = ly * lx                 * vy1 * vx1;
        const int cy0 = min(max(y0, 0), HH - 1), cy1 = min(max(y1, 0), HH - 1);
        const int cx0 = min(max(x0, 0), WW - 1), cx1 = min(max(x1, 0), WW - 1);
        const int base = b * HH * WW;
        pidx[pix][tap][0] = (base + cy0 * WW + cx0) * CC;
        pidx[pix][tap][1] = (base + cy0 * WW + cx1) * CC;
        pidx[pix][tap][2] = (base + cy1 * WW + cx0) * CC;
        pidx[pix][tap][3] = (base + cy1 * WW + cx1) * CC;
    }
    __syncthreads();

    const v8f vzero = {0.f, 0.f, 0.f, 0.f, 0.f, 0.f, 0.f, 0.f};
    v8f acc[2][2];
    acc[0][0] = vzero; acc[0][1] = vzero; acc[1][0] = vzero; acc[1][1] = vzero;

    const int o0   = wv * 32;
    const int apix = tid >> 3;          // pixel this thread fills (0..31)
    const int ac4  = (tid & 7) * 4;     // 4 channels this thread fills

    // Per-wave B pointers (column l16 of each N-subtile, K-half per lane half).
    // B k order used here: lane<16 holds K 0..15, lane>=16 holds K 16..31 (contig, same as A).
    const _Float16* bcol0 = wP + (size_t)(o0 + l16) * KDIM + (khalf * 16);
    const _Float16* bcol1 = bcol0 + (size_t)16 * KDIM;

    for (int tap = 0; tap < KTAPS; ++tap) {
        // hoisted per-tap gather state (pointers constant across the 8 chunks)
        const float4 wt = *(const float4*)&pw[apix][tap][0];
        const int4   ix = *(const int4*)&pidx[apix][tap][0];
        const float* p00 = xT + ix.x + ac4;
        const float* p01 = xT + ix.y + ac4;
        const float* p10 = xT + ix.z + ac4;
        const float* p11 = xT + ix.w + ac4;

        #pragma unroll
        for (int ci = 0; ci < 8; ++ci) {
            const int kk  = tap * CC + ci * 32;
            const int buf = (tap * 8 + ci) & 1;

            // ---- build A chunk: 32 pixels x 32 channels (f16) ----
            {
                const float4 g00 = *(const float4*)(p00 + ci * 32);
                const float4 g01 = *(const float4*)(p01 + ci * 32);
                const float4 g10 = *(const float4*)(p10 + ci * 32);
                const float4 g11 = *(const float4*)(p11 + ci * 32);
                float4 v;
                v.x = fmaf(wt.x, g00.x, fmaf(wt.y, g01.x, fmaf(wt.z, g10.x, wt.w * g11.x)));
                v.y = fmaf(wt.x, g00.y, fmaf(wt.y, g01.y, fmaf(wt.z, g10.y, wt.w * g11.y)));
                v.z = fmaf(wt.x, g00.z, fmaf(wt.y, g01.z, fmaf(wt.z, g10.z, wt.w * g11.z)));
                v.w = fmaf(wt.x, g00.w, fmaf(wt.y, g01.w, fmaf(wt.z, g10.w, wt.w * g11.w)));
                _Float16 hp[4];
                hp[0] = (_Float16)v.x; hp[1] = (_Float16)v.y;
                hp[2] = (_Float16)v.z; hp[3] = (_Float16)v.w;
                *(uint2*)&ldsA[buf][apix * 32 + ac4] = *(const uint2*)hp;
            }
            __syncthreads();   // single barrier per K-step (double buffer)

            // ---- fragments ----
            FragH fa[2], fb[2];
            #pragma unroll
            for (int mi = 0; mi < 2; ++mi) {
                const int pix = mi * 16 + l16;
                // A k order used here: lane<16 -> K 0..15; lane>=16 -> K 16..31 (contig, same as B)
                fa[mi].u[0] = *(const uint4*)&ldsA[buf][pix * 32 + khalf * 16];
                fa[mi].u[1] = *(const uint4*)&ldsA[buf][pix * 32 + khalf * 16 + 8];
            }
            fb[0].u[0] = *(const uint4*)(bcol0 + kk);
            fb[0].u[1] = *(const uint4*)(bcol0 + kk + 8);
            fb[1].u[0] = *(const uint4*)(bcol1 + kk);
            fb[1].u[1] = *(const uint4*)(bcol1 + kk + 8);
            __builtin_prefetch(bcol0 + kk + 32, 0, 0);   // speculative: drop-safe at end

            // ---- 4 WMMAs: D = A x B + C ----
            #pragma unroll
            for (int mi = 0; mi < 2; ++mi)
                #pragma unroll
                for (int ni = 0; ni < 2; ++ni)
                    acc[mi][ni] = __builtin_amdgcn_wmma_f32_16x16x32_f16(
                        false, fa[mi].v, false, fb[ni].v,
                        (short)0, acc[mi][ni], false, false);
        }
    }

    // ---- write y tile (layout: yws[m][o], f32) ----
    #pragma unroll
    for (int mi = 0; mi < 2; ++mi)
        #pragma unroll
        for (int ni = 0; ni < 2; ++ni)
            #pragma unroll
            for (int v = 0; v < 8; ++v) {
                const int row = mBase + mi * 16 + v + 8 * khalf;
                const int col = o0 + ni * 16 + l16;
                yws[(size_t)row * COUT + col] = acc[mi][ni][v];
            }
}

// ---------------- Kernel 4: GroupNorm stats ----------------
// 128 blocks = (b, group); stats[0..127]=mean, stats[128..255]=rsqrt(var+eps)
__global__ void k_gn_stats(const float* __restrict__ yws, float* __restrict__ stats) {
    __shared__ float rs[256], rss[256];
    const int b = blockIdx.x >> 5;
    const int g = blockIdx.x & 31;
    const int tid = threadIdx.x;
    float s = 0.f, ss = 0.f;
    for (int pix = tid; pix < HH * WW; pix += 256) {
        const float* p = yws + ((size_t)(b * HH * WW + pix)) * COUT + g * CPG;
        const float4 a0 = *(const float4*)p;
        const float4 a1 = *(const float4*)(p + 4);
        s  += a0.x + a0.y + a0.z + a0.w + a1.x + a1.y + a1.z + a1.w;
        ss += a0.x * a0.x + a0.y * a0.y + a0.z * a0.z + a0.w * a0.w
            + a1.x * a1.x + a1.y * a1.y + a1.z * a1.z + a1.w * a1.w;
    }
    rs[tid] = s; rss[tid] = ss;
    __syncthreads();
    for (int off = 128; off > 0; off >>= 1) {
        if (tid < off) { rs[tid] += rs[tid + off]; rss[tid] += rss[tid + off]; }
        __syncthreads();
    }
    if (tid == 0) {
        const float inv = 1.f / (float)NPG;
        const float mean = rs[0] * inv;
        const float var  = rss[0] * inv - mean * mean;
        stats[blockIdx.x]       = mean;
        stats[128 + blockIdx.x] = rsqrtf(var + EPSV);
    }
}

// ---------------- Kernel 5: normalize + affine + ReLU, NHWC -> NCHW ----------------
__global__ void k_gn_apply(const float* __restrict__ yws, const float* __restrict__ stats,
                           const float* __restrict__ gamma, const float* __restrict__ beta,
                           float* __restrict__ out) {
    __shared__ float tile[64 * 65];
    const int ct = blockIdx.x & 3;
    const int h  = (blockIdx.x >> 2) & 63;
    const int b  = blockIdx.x >> 8;
    const int c0 = ct * 64;
    const int tid = threadIdx.x;
    #pragma unroll 4
    for (int it = 0; it < 16; ++it) {
        const int w = (tid >> 6) + it * 4;
        const int c = tid & 63;
        tile[w * 65 + c] = yws[((size_t)(b * HH * WW) + h * WW + w) * COUT + c0 + c];
    }
    __syncthreads();
    #pragma unroll 4
    for (int it = 0; it < 16; ++it) {
        const int cc = (tid >> 6) + it * 4;
        const int w  = tid & 63;
        const int c  = c0 + cc;
        const int g  = c >> 3;
        const float mean = stats[b * GN_G + g];
        const float rstd = stats[128 + b * GN_G + g];
        float v = (tile[w * 65 + cc] - mean) * rstd * gamma[c] + beta[c];
        out[(((size_t)(b * COUT + c)) * HH + h) * WW + w] = fmaxf(v, 0.f);
    }
}

// ---------------- launch ----------------
extern "C" void kernel_launch(void* const* d_in, const int* in_sizes, int n_in,
                              void* d_out, int out_size, void* d_ws, size_t ws_size,
                              hipStream_t stream) {
    const float* x      = (const float*)d_in[0];
    const float* offset = (const float*)d_in[1];
    const float* weight = (const float*)d_in[2];
    const float* gamma  = (const float*)d_in[3];
    const float* beta   = (const float*)d_in[4];
    float* out = (float*)d_out;

    // workspace layout
    float*     xT = (float*)d_ws;                                    // 16.78 MB
    _Float16*  wP = (_Float16*)(xT + (size_t)BB * HH * WW * CC);     // 1.18 MB
    float*    yws = (float*)((char*)wP + (size_t)COUT * KDIM * sizeof(_Float16)); // 16.78 MB
    float*  stats = yws + (size_t)MTOT * COUT;                       // 256 floats

    k_transpose_x<<<BB * HH * (CC / 64), 256, 0, stream>>>(x, xT);
    k_pack_w<<<(COUT * KDIM + 255) / 256, 256, 0, stream>>>(weight, wP);
    k_deform_gemm<<<MTOT / 32, 256, 0, stream>>>(xT, offset, wP, yws);
    k_gn_stats<<<BB * GN_G, 256, 0, stream>>>(yws, stats);
    k_gn_apply<<<BB * HH * (COUT / 64), 256, 0, stream>>>(yws, stats, gamma, beta, out);
}
